// AttentionEncoderLayer_46557445489088
// MI455X (gfx1250) — matrix operand order repaired and checked by reading, not compiled, not run on hardware
//
#include <hip/hip_runtime.h>
#include <hip/hip_bf16.h>
#include <math.h>

#define Bc 8
#define Sc 1024
#define Dc 512
#define Hc 8
#define HDc 64
#define FFc 2048
#define Mc (Bc * Sc)

typedef __attribute__((ext_vector_type(16))) _Float16 v16h;
typedef __attribute__((ext_vector_type(8)))  _Float16 v8h;
typedef __attribute__((ext_vector_type(8)))  float    v8f;

__device__ __forceinline__ v8f wmma_f16(v16h a, v16h b, v8f c) {
  return __builtin_amdgcn_wmma_f32_16x16x32_f16(false, a, false, b, (short)0, c, false, false);
}

// ---------------------------------------------------------------- elementwise
__global__ void cvt_f32_to_f16(const float* __restrict__ in, _Float16* __restrict__ out, int n) {
  int i = blockIdx.x * 256 + threadIdx.x;
  if (i < n) out[i] = (_Float16)in[i];
}

__global__ void tau_embed_kernel(const float* __restrict__ tau, const float* __restrict__ freq,
                                 _Float16* __restrict__ te) {
  int i = blockIdx.x * 256 + threadIdx.x;  // over Mc*Dc
  int m = i >> 9;        // / Dc
  int d = i & (Dc - 1);
  float t = tau[m] * freq[d >> 1];
  float v = (d & 1) ? cosf(t) : sinf(t);
  te[i] = (_Float16)v;
}

// ---------------------------------------------------------------- GEMM (WMMA)
// C[M,N] = A[M,K] @ W[N,K]^T + bias;  A,W f16 row-major (K contiguous).
#define TM 128
#define TN 128
#define TK 32
#define LSTR 40   // LDS row stride in halfs (80B, 16B-aligned)

__global__ void __launch_bounds__(256)
gemm_f16(const _Float16* __restrict__ A, const _Float16* __restrict__ W,
         const float* __restrict__ bias,
         float* __restrict__ outF, _Float16* __restrict__ outH,
         const float* __restrict__ t1, const float* __restrict__ t2,
         int M, int N, int K, int mode) {
  __shared__ _Float16 lds[2][2][TM * LSTR];
  const int tid  = threadIdx.x;
  const int lane = tid & 31;
  const int wid  = tid >> 5;
  const int hl   = lane >> 4;    // which 16-lane half
  const int l16  = lane & 15;
  const int m0 = blockIdx.x * TM, n0 = blockIdx.y * TN;
  const int wm = wid & 3, wn = wid >> 2;  // 4x2 wave grid -> 32x64 per wave

  v8f acc[2][4];
#pragma unroll
  for (int i = 0; i < 2; i++)
#pragma unroll
    for (int j = 0; j < 4; j++)
#pragma unroll
      for (int e = 0; e < 8; e++) acc[i][j][e] = 0.0f;

  const int KT = K / TK;
  auto stage = [&](int kt, int buf) {
    const _Float16* gA = A + (size_t)m0 * K + (size_t)kt * TK;
    const _Float16* gW = W + (size_t)n0 * K + (size_t)kt * TK;
    _Float16* la = &lds[buf][0][0];
    _Float16* lb = &lds[buf][1][0];
#pragma unroll
    for (int i = 0; i < 2; i++) {
      int c = tid + 256 * i;
      int r = c >> 2, cc = (c & 3) * 8;
      *(v8h*)&la[r * LSTR + cc] = *(const v8h*)&gA[(size_t)r * K + cc];
      *(v8h*)&lb[r * LSTR + cc] = *(const v8h*)&gW[(size_t)r * K + cc];
    }
  };

  stage(0, 0);
  for (int kt = 0; kt < KT; ++kt) {
    __syncthreads();
    if (kt + 1 < KT) stage(kt + 1, (kt + 1) & 1);
    if (kt + 2 < KT) {
      __builtin_prefetch(A + (size_t)m0 * K + (size_t)(kt + 2) * TK, 0, 1);
      __builtin_prefetch(W + (size_t)n0 * K + (size_t)(kt + 2) * TK, 0, 1);
    }
    const _Float16* la = &lds[kt & 1][0][0];
    const _Float16* lb = &lds[kt & 1][1][0];
    v16h af[2], bf[4];
    // A fragment: lane holds row M=l16; K runs {8*hl..} and {16+8*hl..}
#pragma unroll
    for (int i = 0; i < 2; i++) {
      int row = wm * 32 + i * 16 + l16;
      v8h lo = *(const v8h*)&la[row * LSTR + 8 * hl];
      v8h hi = *(const v8h*)&la[row * LSTR + 16 + 8 * hl];
#pragma unroll
      for (int e = 0; e < 8; e++) { af[i][e] = lo[e]; af[i][8 + e] = hi[e]; }
    }
    // B fragment: lane holds col N=l16; K runs {8*hl..} and {16+8*hl..}
#pragma unroll
    for (int j = 0; j < 4; j++) {
      int row = wn * 64 + j * 16 + l16;
      v8h lo = *(const v8h*)&lb[row * LSTR + 8 * hl];
      v8h hi = *(const v8h*)&lb[row * LSTR + 16 + 8 * hl];
#pragma unroll
      for (int e = 0; e < 8; e++) { bf[j][e] = lo[e]; bf[j][8 + e] = hi[e]; }
    }
#pragma unroll
    for (int i = 0; i < 2; i++)
#pragma unroll
      for (int j = 0; j < 4; j++)
        acc[i][j] = wmma_f16(af[i], bf[j], acc[i][j]);
  }

  // Epilogue. C layout: VGPR v -> row = v + 8*hl (+tile), col = l16 (+tile).
#pragma unroll
  for (int i = 0; i < 2; i++) {
#pragma unroll
    for (int j = 0; j < 4; j++) {
      int n = n0 + wn * 64 + j * 16 + l16;
      float bv = bias ? bias[n] : 0.0f;
#pragma unroll
      for (int v = 0; v < 8; v++) {
        int mr = m0 + wm * 32 + i * 16 + hl * 8 + v;
        float val = acc[i][j][v] + bv;
        if (mode == 0) {
          outF[(size_t)mr * N + n] = val;
        } else if (mode == 1) {  // q/k gate-combine -> head-major f16
          size_t idx = (size_t)mr * N + n;
          float q = val * (1.0f + tanhf(t1[idx])) + t2[idx];
          int b = mr >> 10, s = mr & (Sc - 1);
          int h = n >> 6, d = n & (HDc - 1);
          outH[(((size_t)b * Hc + h) * Sc + s) * HDc + d] = (_Float16)q;
        } else if (mode == 2) {  // plain head-major f16 (V)
          int b = mr >> 10, s = mr & (Sc - 1);
          int h = n >> 6, d = n & (HDc - 1);
          outH[(((size_t)b * Hc + h) * Sc + s) * HDc + d] = (_Float16)val;
        } else {                 // exact GELU -> row-major f16
          float g = 0.5f * val * (1.0f + erff(val * 0.70710678118f));
          outH[(size_t)mr * N + n] = (_Float16)g;
        }
      }
    }
  }
}

// ---------------------------------------------------------------- attention
// One block per (b,h, 128-query tile). 8 waves x 16 query rows, flash-style.
__global__ void __launch_bounds__(256)
attn_kernel(const _Float16* __restrict__ qh, const _Float16* __restrict__ kh,
            const _Float16* __restrict__ vh, const float* __restrict__ tau,
            const float* __restrict__ log_slopes, _Float16* __restrict__ ctx_h) {
  __shared__ _Float16 lds_k[32 * 72];    // K chunk [kv][d]
  __shared__ _Float16 lds_vT[64 * 40];   // V chunk transposed [d][kv]
  __shared__ _Float16 lds_p[8][16 * 40]; // per-wave P tile [row][kv]
  const int tid = threadIdx.x, lane = tid & 31, wid = tid >> 5;
  const int hl = lane >> 4, l16 = lane & 15;
  const int bh = blockIdx.y;
  const int b = bh >> 3, h = bh & (Hc - 1);
  const int q0 = blockIdx.x * 128 + wid * 16;

  // Q fragments (HD=64 -> two K-steps of 32), straight from global
  v16h qf[2];
#pragma unroll
  for (int kk = 0; kk < 2; kk++) {
    const _Float16* qr = qh + ((size_t)bh * Sc + q0 + l16) * HDc + kk * 32;
    v8h lo = *(const v8h*)&qr[8 * hl];
    v8h hi = *(const v8h*)&qr[16 + 8 * hl];
#pragma unroll
    for (int e = 0; e < 8; e++) { qf[kk][e] = lo[e]; qf[kk][8 + e] = hi[e]; }
  }
  const float slope = log1pf(expf(log_slopes[h]));
  v8f tq;
#pragma unroll
  for (int v = 0; v < 8; v++) tq[v] = tau[b * Sc + q0 + v + 8 * hl];

  v8f m_, l_, accv[4];
#pragma unroll
  for (int v = 0; v < 8; v++) { m_[v] = -3.0e38f; l_[v] = 0.0f; }
#pragma unroll
  for (int dt = 0; dt < 4; dt++)
#pragma unroll
    for (int v = 0; v < 8; v++) accv[dt][v] = 0.0f;

  for (int kt = 0; kt < Sc / 32; ++kt) {
    __syncthreads();
    {  // stage K chunk and V^T chunk
      int r = tid >> 3, cc = (tid & 7) * 8;
      size_t base = ((size_t)bh * Sc + kt * 32 + r) * HDc + cc;
      *(v8h*)&lds_k[r * 72 + cc] = *(const v8h*)&kh[base];
      v8h vv = *(const v8h*)&vh[base];
#pragma unroll
      for (int e = 0; e < 8; e++) lds_vT[(cc + e) * 40 + r] = vv[e];
    }
    __syncthreads();

    // logits: two 16x16 tiles over this 32-wide kv chunk
    v8f s[2];
#pragma unroll
    for (int nt = 0; nt < 2; nt++) {
#pragma unroll
      for (int v = 0; v < 8; v++) s[nt][v] = 0.0f;
#pragma unroll
      for (int kk = 0; kk < 2; kk++) {
        int row = nt * 16 + l16;
        v8h lo = *(const v8h*)&lds_k[row * 72 + kk * 32 + 8 * hl];
        v8h hi = *(const v8h*)&lds_k[row * 72 + kk * 32 + 16 + 8 * hl];
        v16h bf;
#pragma unroll
        for (int e = 0; e < 8; e++) { bf[e] = lo[e]; bf[8 + e] = hi[e]; }
        s[nt] = wmma_f16(qf[kk], bf, s[nt]);
      }
      int kcol = kt * 32 + nt * 16 + l16;
      float tk = tau[b * Sc + kcol];
#pragma unroll
      for (int v = 0; v < 8; v++)
        s[nt][v] = s[nt][v] * 0.125f - slope * fabsf(tq[v] - tk);
    }

    // online softmax: row reductions over 16-lane groups
    v8f mc;
#pragma unroll
    for (int v = 0; v < 8; v++) mc[v] = fmaxf(s[0][v], s[1][v]);
    for (int mask = 1; mask < 16; mask <<= 1)
#pragma unroll
      for (int v = 0; v < 8; v++) mc[v] = fmaxf(mc[v], __shfl_xor(mc[v], mask, 32));
    v8f mn, p0, p1, lc, corr;
#pragma unroll
    for (int v = 0; v < 8; v++) {
      mn[v] = fmaxf(m_[v], mc[v]);
      p0[v] = expf(s[0][v] - mn[v]);
      p1[v] = expf(s[1][v] - mn[v]);
      lc[v] = p0[v] + p1[v];
      corr[v] = expf(m_[v] - mn[v]);
    }
    for (int mask = 1; mask < 16; mask <<= 1)
#pragma unroll
      for (int v = 0; v < 8; v++) lc[v] += __shfl_xor(lc[v], mask, 32);
#pragma unroll
    for (int v = 0; v < 8; v++) {
      l_[v] = l_[v] * corr[v] + lc[v];
      m_[v] = mn[v];
    }
#pragma unroll
    for (int dt = 0; dt < 4; dt++)
#pragma unroll
      for (int v = 0; v < 8; v++) accv[dt][v] *= corr[v];

    // C-layout P -> LDS -> A-fragment layout
    _Float16* P = lds_p[wid];
#pragma unroll
    for (int v = 0; v < 8; v++) {
      int row = v + 8 * hl;
      P[row * 40 + l16]      = (_Float16)p0[v];
      P[row * 40 + 16 + l16] = (_Float16)p1[v];
    }
    __syncthreads();
    v16h pf;
    {
      v8h lo = *(const v8h*)&P[l16 * 40 + 8 * hl];
      v8h hi = *(const v8h*)&P[l16 * 40 + 16 + 8 * hl];
#pragma unroll
      for (int e = 0; e < 8; e++) { pf[e] = lo[e]; pf[8 + e] = hi[e]; }
    }
    // P(16x32) @ V(32x64) : 4 WMMAs
#pragma unroll
    for (int dt = 0; dt < 4; dt++) {
      v8h lo = *(const v8h*)&lds_vT[(dt * 16 + l16) * 40 + 8 * hl];
      v8h hi = *(const v8h*)&lds_vT[(dt * 16 + l16) * 40 + 16 + 8 * hl];
      v16h bf;
#pragma unroll
      for (int e = 0; e < 8; e++) { bf[e] = lo[e]; bf[8 + e] = hi[e]; }
      accv[dt] = wmma_f16(pf, bf, accv[dt]);
    }
  }

  // normalize + store token-major f16 ctx
#pragma unroll
  for (int dt = 0; dt < 4; dt++)
#pragma unroll
    for (int v = 0; v < 8; v++) {
      int row = q0 + v + 8 * hl;
      int col = h * HDc + dt * 16 + l16;
      ctx_h[((size_t)b * Sc + row) * Dc + col] = (_Float16)(accv[dt][v] / l_[v]);
    }
}

// ---------------------------------------------------------------- layernorm
__global__ void __launch_bounds__(256)
ln_kernel(const float* __restrict__ a, const float* __restrict__ r,
          const float* __restrict__ g, const float* __restrict__ be,
          float* __restrict__ outF, _Float16* __restrict__ outH) {
  __shared__ float red[256];
  int m = blockIdx.x, tid = threadIdx.x;
  size_t base = (size_t)m * Dc;
  float x0 = a[base + tid] + r[base + tid];
  float x1 = a[base + tid + 256] + r[base + tid + 256];
  red[tid] = x0 + x1;
  __syncthreads();
  for (int off = 128; off > 0; off >>= 1) {
    if (tid < off) red[tid] += red[tid + off];
    __syncthreads();
  }
  float mean = red[0] * (1.0f / Dc);
  __syncthreads();
  float d0 = x0 - mean, d1 = x1 - mean;
  red[tid] = d0 * d0 + d1 * d1;
  __syncthreads();
  for (int off = 128; off > 0; off >>= 1) {
    if (tid < off) red[tid] += red[tid + off];
    __syncthreads();
  }
  float inv = rsqrtf(red[0] * (1.0f / Dc) + 1e-5f);
  float y0 = d0 * inv * g[tid] + be[tid];
  float y1 = d1 * inv * g[tid + 256] + be[tid + 256];
  outF[base + tid] = y0;
  outF[base + tid + 256] = y1;
  if (outH) {
    outH[base + tid] = (_Float16)y0;
    outH[base + tid + 256] = (_Float16)y1;
  }
}

// ---------------------------------------------------------------- launch
static inline void launch_gemm(const _Float16* A, const _Float16* W, const float* bias,
                               float* outF, _Float16* outH, const float* t1, const float* t2,
                               int M, int N, int K, int mode, hipStream_t stream) {
  dim3 grid(M / TM, N / TN);
  gemm_f16<<<grid, 256, 0, stream>>>(A, W, bias, outF, outH, t1, t2, M, N, K, mode);
}

extern "C" void kernel_launch(void* const* d_in, const int* in_sizes, int n_in,
                              void* d_out, int out_size, void* d_ws, size_t ws_size,
                              hipStream_t stream) {
  (void)in_sizes; (void)n_in; (void)out_size; (void)ws_size;
  const float* src        = (const float*)d_in[0];
  const float* tau        = (const float*)d_in[1];
  const float* Wq  = (const float*)d_in[2];  const float* bq  = (const float*)d_in[3];
  const float* Wk  = (const float*)d_in[4];  const float* bk  = (const float*)d_in[5];
  const float* Wv  = (const float*)d_in[6];  const float* bv  = (const float*)d_in[7];
  const float* Wo  = (const float*)d_in[8];  const float* bo  = (const float*)d_in[9];
  const float* Wqg = (const float*)d_in[10]; const float* bqg = (const float*)d_in[11];
  const float* Wqb = (const float*)d_in[12]; const float* bqb = (const float*)d_in[13];
  const float* Wkg = (const float*)d_in[14]; const float* bkg = (const float*)d_in[15];
  const float* Wkb = (const float*)d_in[16]; const float* bkb = (const float*)d_in[17];
  const float* log_slopes = (const float*)d_in[18];
  const float* W1  = (const float*)d_in[19]; const float* b1  = (const float*)d_in[20];
  const float* W2  = (const float*)d_in[21]; const float* b2  = (const float*)d_in[22];
  const float* ln1_g = (const float*)d_in[23]; const float* ln1_b = (const float*)d_in[24];
  const float* ln2_g = (const float*)d_in[25]; const float* ln2_b = (const float*)d_in[26];
  const float* tau_freq = (const float*)d_in[27];
  float* out = (float*)d_out;

  char* ws = (char*)d_ws;
  size_t off = 0;
  auto alloc = [&](size_t bytes) { size_t o = off; off += (bytes + 255) & ~(size_t)255; return o; };
  const size_t DD = (size_t)Dc * Dc, MD = (size_t)Mc * Dc, FD = (size_t)FFc * Dc;
  _Float16* Wq_h  = (_Float16*)(ws + alloc(DD * 2));
  _Float16* Wk_h  = (_Float16*)(ws + alloc(DD * 2));
  _Float16* Wv_h  = (_Float16*)(ws + alloc(DD * 2));
  _Float16* Wo_h  = (_Float16*)(ws + alloc(DD * 2));
  _Float16* Wqg_h = (_Float16*)(ws + alloc(DD * 2));
  _Float16* Wqb_h = (_Float16*)(ws + alloc(DD * 2));
  _Float16* Wkg_h = (_Float16*)(ws + alloc(DD * 2));
  _Float16* Wkb_h = (_Float16*)(ws + alloc(DD * 2));
  _Float16* W1_h  = (_Float16*)(ws + alloc(FD * 2));
  _Float16* W2_h  = (_Float16*)(ws + alloc(FD * 2));
  _Float16* src_h = (_Float16*)(ws + alloc(MD * 2));
  _Float16* te_h  = (_Float16*)(ws + alloc(MD * 2));
  _Float16* qh    = (_Float16*)(ws + alloc(MD * 2));
  _Float16* khb   = (_Float16*)(ws + alloc(MD * 2));
  _Float16* vhb   = (_Float16*)(ws + alloc(MD * 2));
  _Float16* ctx_h = (_Float16*)(ws + alloc(MD * 2));
  _Float16* x_h   = (_Float16*)(ws + alloc(MD * 2));
  _Float16* ff1_h = (_Float16*)(ws + alloc((size_t)Mc * FFc * 2));
  float* tmp1 = (float*)(ws + alloc(MD * 4));
  float* tmp2 = (float*)(ws + alloc(MD * 4));

  // 1) weights + activations to f16; tau embedding
  cvt_f32_to_f16<<<DD / 256, 256, 0, stream>>>(Wq, Wq_h, (int)DD);
  cvt_f32_to_f16<<<DD / 256, 256, 0, stream>>>(Wk, Wk_h, (int)DD);
  cvt_f32_to_f16<<<DD / 256, 256, 0, stream>>>(Wv, Wv_h, (int)DD);
  cvt_f32_to_f16<<<DD / 256, 256, 0, stream>>>(Wo, Wo_h, (int)DD);
  cvt_f32_to_f16<<<DD / 256, 256, 0, stream>>>(Wqg, Wqg_h, (int)DD);
  cvt_f32_to_f16<<<DD / 256, 256, 0, stream>>>(Wqb, Wqb_h, (int)DD);
  cvt_f32_to_f16<<<DD / 256, 256, 0, stream>>>(Wkg, Wkg_h, (int)DD);
  cvt_f32_to_f16<<<DD / 256, 256, 0, stream>>>(Wkb, Wkb_h, (int)DD);
  cvt_f32_to_f16<<<FD / 256, 256, 0, stream>>>(W1, W1_h, (int)FD);
  cvt_f32_to_f16<<<FD / 256, 256, 0, stream>>>(W2, W2_h, (int)FD);
  cvt_f32_to_f16<<<MD / 256, 256, 0, stream>>>(src, src_h, (int)MD);
  tau_embed_kernel<<<MD / 256, 256, 0, stream>>>(tau, tau_freq, te_h);

  // 2) q/k/v with tau gating
  launch_gemm(te_h, Wqg_h, bqg, tmp1, nullptr, nullptr, nullptr, Mc, Dc, Dc, 0, stream);
  launch_gemm(te_h, Wqb_h, bqb, tmp2, nullptr, nullptr, nullptr, Mc, Dc, Dc, 0, stream);
  launch_gemm(src_h, Wq_h, bq, nullptr, qh, tmp1, tmp2, Mc, Dc, Dc, 1, stream);
  launch_gemm(te_h, Wkg_h, bkg, tmp1, nullptr, nullptr, nullptr, Mc, Dc, Dc, 0, stream);
  launch_gemm(te_h, Wkb_h, bkb, tmp2, nullptr, nullptr, nullptr, Mc, Dc, Dc, 0, stream);
  launch_gemm(src_h, Wk_h, bk, nullptr, khb, tmp1, tmp2, Mc, Dc, Dc, 1, stream);
  launch_gemm(src_h, Wv_h, bv, nullptr, vhb, nullptr, nullptr, Mc, Dc, Dc, 2, stream);

  // 3) flash attention with relative-tau bias
  attn_kernel<<<dim3(Sc / 128, Bc * Hc), 256, 0, stream>>>(qh, khb, vhb, tau, log_slopes, ctx_h);

  // 4) output projection + LN1
  launch_gemm(ctx_h, Wo_h, bo, tmp1, nullptr, nullptr, nullptr, Mc, Dc, Dc, 0, stream);
  ln_kernel<<<Mc, 256, 0, stream>>>(src, tmp1, ln1_g, ln1_b, tmp2, x_h);

  // 5) FFN (GELU) + LN2
  launch_gemm(x_h, W1_h, b1, nullptr, ff1_h, nullptr, nullptr, Mc, FFc, Dc, 3, stream);
  launch_gemm(ff1_h, W2_h, b2, tmp1, nullptr, nullptr, nullptr, Mc, Dc, FFc, 0, stream);
  ln_kernel<<<Mc, 256, 0, stream>>>(tmp2, tmp1, ln2_g, ln2_b, out, nullptr);
}
